// SinkhornKnopp_45423574122874
// MI455X (gfx1250) — matrix operand order repaired and checked by reading, not compiled, not run on hardware
//
#include <hip/hip_runtime.h>
#include <hip/hip_bf16.h>
#include <math.h>

// Sinkhorn-Knopp, 8192x8192.
// Strategy: Q stored fp16 (128 MB -> L2-resident on MI455X's 192 MB L2);
// normalization of Q provably cancels in final T; iterations are two GEMVs.
// Row-sum GEMV uses v_wmma_f32_16x16x32_f16.

typedef _Float16 half8  __attribute__((ext_vector_type(8)));
typedef _Float16 half4  __attribute__((ext_vector_type(4)));
typedef _Float16 half2v __attribute__((ext_vector_type(2)));
typedef _Float16 v16h   __attribute__((ext_vector_type(16)));
typedef float    v8f    __attribute__((ext_vector_type(8)));

#define NDIM 8192
#define MDIM 8192
#define STAT_BLOCKS 2048
#define VSCALE 8192.0f

// ---------------- init ----------------
__global__ void init_kernel(float* __restrict__ U, float* __restrict__ colsum,
                            float* __restrict__ rowsum, int n) {
  int i = blockIdx.x * blockDim.x + threadIdx.x;
  if (i < n) { U[i] = 1.0f; colsum[i] = 0.0f; rowsum[i] = 0.0f; }
}

// ---------------- pass 1: min/sum/sumsq partials ----------------
__global__ void stats_partial(const float* __restrict__ C, size_t n,
                              float2* __restrict__ psum, unsigned* __restrict__ pmin) {
  __shared__ float ssum[256], ssq[256];
  __shared__ unsigned smin[256];
  size_t stride = (size_t)gridDim.x * blockDim.x * 4;
  size_t start  = ((size_t)blockIdx.x * blockDim.x + threadIdx.x) * 4;
  float lsum = 0.0f, lsq = 0.0f, lmin = 3.4e38f;
  for (size_t i = start; i < n; i += stride) {
    float4 v = *(const float4*)(C + i);
    lsum += (v.x + v.y) + (v.z + v.w);
    lsq  += (v.x * v.x + v.y * v.y) + (v.z * v.z + v.w * v.w);
    lmin = fminf(lmin, fminf(fminf(v.x, v.y), fminf(v.z, v.w)));
  }
  ssum[threadIdx.x] = lsum;
  ssq[threadIdx.x]  = lsq;
  smin[threadIdx.x] = __float_as_uint(lmin);  // valid: all inputs >= 0
  __syncthreads();
  for (int off = 128; off > 0; off >>= 1) {
    if ((int)threadIdx.x < off) {
      ssum[threadIdx.x] += ssum[threadIdx.x + off];
      ssq[threadIdx.x]  += ssq[threadIdx.x + off];
      smin[threadIdx.x] = min(smin[threadIdx.x], smin[threadIdx.x + off]);
    }
    __syncthreads();
  }
  if (threadIdx.x == 0) {
    psum[blockIdx.x] = make_float2(ssum[0], ssq[0]);
    pmin[blockIdx.x] = smin[0];
  }
}

// deterministic single-block finalize -> stats[0]=min, stats[1]=1/(std*TEMP)
__global__ void stats_finalize(const float2* __restrict__ psum,
                               const unsigned* __restrict__ pmin,
                               int nparts, float* __restrict__ stats, double n) {
  __shared__ double dsum[256], dsq[256];
  __shared__ unsigned smin[256];
  double ls = 0.0, lq = 0.0; unsigned lm = 0x7f800000u;
  for (int i = threadIdx.x; i < nparts; i += 256) {
    float2 p = psum[i];
    ls += (double)p.x; lq += (double)p.y;
    lm = min(lm, pmin[i]);
  }
  dsum[threadIdx.x] = ls; dsq[threadIdx.x] = lq; smin[threadIdx.x] = lm;
  __syncthreads();
  for (int off = 128; off > 0; off >>= 1) {
    if ((int)threadIdx.x < off) {
      dsum[threadIdx.x] += dsum[threadIdx.x + off];
      dsq[threadIdx.x]  += dsq[threadIdx.x + off];
      smin[threadIdx.x] = min(smin[threadIdx.x], smin[threadIdx.x + off]);
    }
    __syncthreads();
  }
  if (threadIdx.x == 0) {
    double var = (dsq[0] - dsum[0] * dsum[0] / n) / (n - 1.0);  // ddof=1
    stats[0] = __uint_as_float(smin[0]);
    stats[1] = (float)(1.0 / (sqrt(var) * 0.2));  // TEMP = 0.2
  }
}

// ---------------- pass 2: Q = exp(-(c-min)*inv) stored fp16 ----------------
__global__ void expq_kernel(const float* __restrict__ C, _Float16* __restrict__ Q,
                            const float* __restrict__ stats, size_t n) {
  float mn = stats[0], a = stats[1];
  size_t stride = (size_t)gridDim.x * blockDim.x * 4;
  for (size_t i = 4 * ((size_t)blockIdx.x * blockDim.x + threadIdx.x); i < n; i += stride) {
    float4 v = *(const float4*)(C + i);
    half4 q;
    q[0] = (_Float16)__expf((mn - v.x) * a);
    q[1] = (_Float16)__expf((mn - v.y) * a);
    q[2] = (_Float16)__expf((mn - v.z) * a);
    q[3] = (_Float16)__expf((mn - v.w) * a);
    *(half4*)(Q + i) = q;
  }
}

// ---------------- colsum[j] = sum_i U[i]*Q[i,j]  (coalesced VALU) ----------------
__global__ void colsum_kernel(const _Float16* __restrict__ Q, const float* __restrict__ U,
                              float* __restrict__ colsum) {
  int j2 = blockIdx.x * blockDim.x + threadIdx.x;  // column pair
  int i0 = blockIdx.y * 256;
  float a0 = 0.0f, a1 = 0.0f;
  const _Float16* p = Q + (size_t)i0 * MDIM + 2 * j2;
  for (int i = 0; i < 256; ++i) {
    half2v q = *(const half2v*)p;
    float u = U[i0 + i];
    a0 += u * (float)q[0];
    a1 += u * (float)q[1];
    p += MDIM;
  }
  atomicAdd(&colsum[2 * j2 + 0], a0);
  atomicAdd(&colsum[2 * j2 + 1], a1);
}

// V[j] = B[j]/colsum[j]; Vh = fp16(V*8192) keeps values in fp16 normal range
__global__ void vupdate(const float* __restrict__ Bv, const float* __restrict__ colsum,
                        float* __restrict__ V, _Float16* __restrict__ Vh,
                        float* __restrict__ rowsum, int m) {
  int j = blockIdx.x * blockDim.x + threadIdx.x;
  if (j < m) {
    float v = Bv[j] / colsum[j];
    V[j] = v;
    Vh[j] = (_Float16)(v * VSCALE);
    rowsum[j] = 0.0f;  // zero before accumulation this iteration
  }
}

// ---------------- rowsum[i] = sum_j Q[i,j]*Vh[j] via WMMA ----------------
// A fragment (16x32 f16): lane L row = L&15,
//   halves 0..7  -> K = (L<16?0:16) + e   -> contiguous b128
//   halves 8..15 -> K = (L<16?8:24) + e-8 -> contiguous b128 (+8 elems)
// B fragment (32x16 f16): lane L col = L&15, halves e -> K = (L<16?0:16)+e,
//   replicated across all 16 columns (same chunk of Vh) -> every D column
//   holds the partial row sums.
__global__ void rowsum_wmma(const _Float16* __restrict__ Q, const _Float16* __restrict__ Vh,
                            float* __restrict__ rowsum) {
  const int lane  = threadIdx.x & 31;
  const int wave  = threadIdx.x >> 5;              // 4 waves / block
  const int strip = blockIdx.x * 4 + wave;         // 16-row strip, 0..511
  const int rbase = strip * 16;
  const int hi    = lane >> 4;                     // half-wave selector
  const int r     = rbase + (lane & 15);
  const size_t rowoff = (size_t)r * MDIM;
  const int jbase0 = blockIdx.y * 1024;            // K-split: 8 chunks of 1024

  v8f c = {0.0f, 0.0f, 0.0f, 0.0f, 0.0f, 0.0f, 0.0f, 0.0f};
  for (int j = jbase0; j < jbase0 + 1024; j += 32) {
    const _Float16* pa = Q + rowoff + j + hi * 16;
    half8 a0 = *(const half8*)(pa);
    half8 a1 = *(const half8*)(pa + 8);
    v16h a = __builtin_shufflevector(a0, a1, 0, 1, 2, 3, 4, 5, 6, 7,
                                             8, 9, 10, 11, 12, 13, 14, 15);
    v16h b = *(const v16h*)(Vh + j + hi * 16);
    c = __builtin_amdgcn_wmma_f32_16x16x32_f16(false, a, false, b,
                                               (short)0, c, false, false);
  }
  // D layout: VGPR v -> row rbase + v + (hi?8:0); columns identical -> read n==0
  if ((lane & 15) == 0) {
    int base = rbase + (hi ? 8 : 0);
#pragma unroll
    for (int v = 0; v < 8; ++v) atomicAdd(&rowsum[base + v], c[v]);
  }
}

// U[i] = A[i]*8192/rowsum_wmma[i]  (undo the Vh scale); zero colsum for next iter
__global__ void uupdate(const float* __restrict__ Av, const float* __restrict__ rowsum,
                        float* __restrict__ U, float* __restrict__ colsum, int n) {
  int i = blockIdx.x * blockDim.x + threadIdx.x;
  if (i < n) {
    U[i] = Av[i] * VSCALE / rowsum[i];
    colsum[i] = 0.0f;
  }
}

// ---------------- T = U[:,None] * Q * V[None,:] ----------------
__global__ void final_kernel(const _Float16* __restrict__ Q, const float* __restrict__ U,
                             const float* __restrict__ V, float* __restrict__ T) {
  size_t e = 4 * ((size_t)blockIdx.x * blockDim.x + threadIdx.x);
  int i = (int)(e >> 13);
  int j = (int)(e & (MDIM - 1));
  half4 q = *(const half4*)(Q + e);
  float u = U[i];
  float4 v = *(const float4*)(V + j);
  float4 t;
  t.x = u * (float)q[0] * v.x;
  t.y = u * (float)q[1] * v.y;
  t.z = u * (float)q[2] * v.z;
  t.w = u * (float)q[3] * v.w;
  *(float4*)(T + e) = t;
}

extern "C" void kernel_launch(void* const* d_in, const int* in_sizes, int n_in,
                              void* d_out, int out_size, void* d_ws, size_t ws_size,
                              hipStream_t stream) {
  const float* cdist = (const float*)d_in[0];
  const float* Av    = (const float*)d_in[1];
  const float* Bv    = (const float*)d_in[2];
  float* T = (float*)d_out;

  constexpr size_t NM = (size_t)NDIM * MDIM;
  char* ws = (char*)d_ws;
  _Float16* Q = (_Float16*)ws;
  size_t off = (NM * sizeof(_Float16) + 255) & ~(size_t)255;
  float*    U      = (float*)(ws + off);     off += NDIM * sizeof(float);
  float*    V      = (float*)(ws + off);     off += MDIM * sizeof(float);
  _Float16* Vh     = (_Float16*)(ws + off);  off += MDIM * sizeof(_Float16);
  float*    rowsum = (float*)(ws + off);     off += NDIM * sizeof(float);
  float*    colsum = (float*)(ws + off);     off += MDIM * sizeof(float);
  float2*   psum   = (float2*)(ws + off);    off += STAT_BLOCKS * sizeof(float2);
  unsigned* pmin   = (unsigned*)(ws + off);  off += STAT_BLOCKS * sizeof(unsigned);
  float*    stats  = (float*)(ws + off);     off += 2 * sizeof(float);

  init_kernel<<<NDIM / 256, 256, 0, stream>>>(U, colsum, rowsum, NDIM);
  stats_partial<<<STAT_BLOCKS, 256, 0, stream>>>(cdist, NM, psum, pmin);
  stats_finalize<<<1, 256, 0, stream>>>(psum, pmin, STAT_BLOCKS, stats, (double)NM);
  expq_kernel<<<4096, 256, 0, stream>>>(cdist, Q, stats, NM);

  for (int it = 0; it < 10; ++it) {
    colsum_kernel<<<dim3(MDIM / 512, 32), 256, 0, stream>>>(Q, U, colsum);
    vupdate<<<MDIM / 256, 256, 0, stream>>>(Bv, colsum, V, Vh, rowsum, MDIM);
    rowsum_wmma<<<dim3(128, 8), 128, 0, stream>>>(Q, Vh, rowsum);
    uupdate<<<NDIM / 256, 256, 0, stream>>>(Av, rowsum, U, colsum, NDIM);
  }

  final_kernel<<<(unsigned)(NM / 1024), 256, 0, stream>>>(Q, U, V, T);
}
